// RNNModel_67250597921293
// MI455X (gfx1250) — matrix operand order repaired and checked by reading, not compiled, not run on hardware
//
#include <hip/hip_runtime.h>
#include <hip/hip_bf16.h>

// ---------------------------------------------------------------------------
// RNN LM on gfx1250 (CDNA5, wave32):
//   k1: embedding gather + bias           (memory-bound)
//   k2: batch-parallel recurrence, W_hh in LDS; emits Y as split f16 hi/lo
//   k2b: one-shot W_out f32 -> f16 hi/lo split
//   k3: logits GEMM, TDM (tensor_load_to_lds) staging into padded LDS,
//       v_wmma_f32_16x16x32_f16 with 3-term hi/lo split (near-fp32 accuracy)
// ---------------------------------------------------------------------------

#define VOCAB  10000
#define HIDDEN 256
#define BATCH  32
#define SEQ    256
#define ROWS   (SEQ * BATCH)          // 8192 rows of Y / logits

typedef __attribute__((ext_vector_type(16))) _Float16 v16h;
typedef __attribute__((ext_vector_type(8)))  _Float16 v8h;
typedef __attribute__((ext_vector_type(4)))  _Float16 v4h;
typedef __attribute__((ext_vector_type(8)))  float    v8f;
typedef __attribute__((ext_vector_type(4)))  unsigned int v4u;
typedef __attribute__((ext_vector_type(8)))  int      v8i;
typedef __attribute__((ext_vector_type(4)))  int      v4i;

union AFrag { v16h v; v4h q[4]; };
union BFrag { v16h v; v8h o[2]; };

#if defined(__has_builtin)
#if __has_builtin(__builtin_amdgcn_tensor_load_to_lds)
#define USE_TDM 1
#endif
#endif

// ---------------------------------------------------------------------------
// Kernel 1: xproj[t][b][i] = W_ih[i][tok(b,t)] + b_ih[i] + b_hh[i]
// grid = 8192 blocks (t*32+b), block = 256 threads (i)
// ---------------------------------------------------------------------------
__global__ void rnn_gather(const int* __restrict__ inputs,
                           const float* __restrict__ W_ih,
                           const float* __restrict__ b_ih,
                           const float* __restrict__ b_hh,
                           float* __restrict__ xproj) {
  const int gid = blockIdx.x;          // t*BATCH + b
  const int t   = gid >> 5;
  const int b   = gid & 31;
  const int i   = threadIdx.x;
  const int tok = inputs[b * SEQ + t];                 // inputs is (BATCH, SEQ)
  const float v = W_ih[(size_t)i * VOCAB + tok] + b_ih[i] + b_hh[i];
  xproj[(size_t)gid * HIDDEN + i] = v;
}

// ---------------------------------------------------------------------------
// Kernel 2: recurrence, one workgroup per batch row (rows are independent).
// W_hh in LDS, row stride 260 dwords (bank-spread, 16B aligned), double-
// buffered h, one barrier/step. Y emitted pre-split into f16 hi/lo planes.
// dyn LDS = (256*260 + 2*256) * 4 = 268288 B
// ---------------------------------------------------------------------------
#define WSTRIDE 260
__global__ void rnn_recurrence(const float* __restrict__ xproj,
                               const float* __restrict__ state,
                               const float* __restrict__ W_hh,
                               _Float16* __restrict__ Yh,
                               _Float16* __restrict__ Yl,
                               float* __restrict__ h_last) {
  extern __shared__ float lds[];
  float* W    = lds;                        // [256][WSTRIDE]
  float* hbuf = lds + HIDDEN * WSTRIDE;     // [2][256]

  const int b   = blockIdx.x;
  const int tid = threadIdx.x;              // 0..255, output index i

  for (int idx = tid; idx < HIDDEN * HIDDEN; idx += 256) {
    const int i = idx >> 8;
    const int j = idx & 255;
    W[i * WSTRIDE + j] = W_hh[idx];
  }
  hbuf[tid] = state[b * HIDDEN + tid];
  __syncthreads();

  int cur = 0;
  float hn = 0.0f;
  for (int t = 0; t < SEQ; ++t) {
    float acc = xproj[((size_t)t * BATCH + b) * HIDDEN + tid];
    const float* hc = hbuf + cur * HIDDEN;
    const float* wr = W + tid * WSTRIDE;    // byte base = 1040*tid, 16B aligned
#pragma unroll 8
    for (int j = 0; j < HIDDEN; j += 4) {
      const float4 hv = *reinterpret_cast<const float4*>(hc + j);  // broadcast
      const float4 wv = *reinterpret_cast<const float4*>(wr + j);  // b128
      acc = fmaf(hv.x, wv.x, acc);
      acc = fmaf(hv.y, wv.y, acc);
      acc = fmaf(hv.z, wv.z, acc);
      acc = fmaf(hv.w, wv.w, acc);
    }
    hn = tanhf(acc);
    const size_t yi = ((size_t)t * BATCH + b) * HIDDEN + tid;
    const _Float16 hh = (_Float16)hn;
    Yh[yi] = hh;
    Yl[yi] = (_Float16)(hn - (float)hh);
    hbuf[(cur ^ 1) * HIDDEN + tid] = hn;
    __syncthreads();
    cur ^= 1;
  }
  h_last[b * HIDDEN + tid] = hn;
}

// ---------------------------------------------------------------------------
// Kernel 2b: one-shot split of W_out (10000x256 f32) into f16 hi/lo planes.
// grid = 2500 blocks x 256 threads, float4 per thread.
// ---------------------------------------------------------------------------
__global__ void wout_split(const float* __restrict__ W_out,
                           _Float16* __restrict__ Wh,
                           _Float16* __restrict__ Wl) {
  const size_t idx = ((size_t)blockIdx.x * 256 + threadIdx.x) * 4;
  const float4 v = *reinterpret_cast<const float4*>(W_out + idx);
  const _Float16 h0 = (_Float16)v.x, h1 = (_Float16)v.y,
                 h2 = (_Float16)v.z, h3 = (_Float16)v.w;
  v4h hi = {h0, h1, h2, h3};
  v4h lo = {(_Float16)(v.x - (float)h0), (_Float16)(v.y - (float)h1),
            (_Float16)(v.z - (float)h2), (_Float16)(v.w - (float)h3)};
  *reinterpret_cast<v4h*>(Wh + idx) = hi;
  *reinterpret_cast<v4h*>(Wl + idx) = lo;
}

// ---------------------------------------------------------------------------
// Kernel 3: logits = Y @ W_out^T + b_out   (8192x256 @ 256x10000)
// Block = 128(M) x 80(N), full K=256 in LDS, 8 waves, wave w owns M-rows
// [16w,16w+16) x 5 N-tiles. grid = (64, 125).
// LDS f16 row stride 264 (dword stride 132 -> bank-spread; 16B aligned).
// Staging via TDM: 4x tensor_load_to_lds with hardware row padding
// (pad_interval=6 -> every 128 dwords (one 512B row), pad_amount=3 -> 4
// dwords = 8 f16), reproducing the 264-element row stride. TENSORcnt waits.
// dyn LDS = (128+80)*264*2 planes *2B = 219648 B
// ---------------------------------------------------------------------------
#define BM 128
#define BN 80
#define NT 5
#define LSTR 264                         // f16 elements per LDS row
#define A_SZ (BM * LSTR * 2)             // 67584 bytes per A plane
#define B_SZ (BN * LSTR * 2)             // 42240 bytes per B plane

#ifdef USE_TDM
// One 2-D f16 tile (rows x 256, row-major, contiguous) -> LDS with padding.
// D# packing per CDNA5 ISA 8.3/8.4 (group0: count/lds/global/type,
// group1: data_size, pad ctl, tensor/tile dims, dim0 stride).
// This toolchain's builtin takes 6 args: (g0 v4u, g1 v8i, g2 v4i, g3 v4i,
// extra v8i, cpol) -- groups 2/3 and the trailing group are zero for 2-D.
__device__ __forceinline__ void tdm_load_tile_f16(unsigned int lds_off,
                                                  const _Float16* g,
                                                  unsigned int rows) {
  const unsigned long long ga = (unsigned long long)(uintptr_t)g;
  v4u g0;
  g0[0] = 1u;                                        // count=1, user mode
  g0[1] = lds_off;                                   // lds_addr (bytes)
  g0[2] = (unsigned int)ga;                          // global_addr[31:0]
  g0[3] = (unsigned int)((ga >> 32) & 0x01FFFFFFu)   // global_addr[56:32]
        | (2u << 30);                                // type=2 (image)
  v8i g1;
  g1[0] = (int)((1u << 16)      // data_size = 1 -> 2 bytes
              | (1u << 20)      // pad_enable
              | (6u << 22)      // pad_interval: 128 dwords (one 512B row)
              | (3u << 25));    // pad_amount: 4 dwords (8 f16)
  g1[1] = (int)(256u << 16);    // tensor_dim0[15:0]=256 (elements)
  g1[2] = (int)(rows << 16);    // tensor_dim0[31:16]=0 | tensor_dim1[15:0]
  g1[3] = (int)(256u << 16);    // tensor_dim1[31:16]=0 | tile_dim0=256
  g1[4] = (int)rows;            // tile_dim1=rows | tile_dim2=0
  g1[5] = 256;                  // tensor_dim0_stride[31:0] = 256 elements
  g1[6] = 0;                    // stride hi | tensor_dim1_stride lo (unused)
  g1[7] = 0;
  const v4i z4 = {0, 0, 0, 0};
  const v8i z8 = {0, 0, 0, 0, 0, 0, 0, 0};
  __builtin_amdgcn_tensor_load_to_lds(g0, g1, z4, z4, z8, 0);
}
#endif

__global__ void rnn_logits_gemm(const _Float16* __restrict__ Yh,
                                const _Float16* __restrict__ Yl,
                                const _Float16* __restrict__ Wh,
                                const _Float16* __restrict__ Wl,
                                const float* __restrict__ b_out,
                                float* __restrict__ logits) {
  extern __shared__ _Float16 sh[];
  _Float16* Ah = sh;                                   // [BM][LSTR]
  _Float16* Al = Ah + BM * LSTR;
  _Float16* Bh = Al + BM * LSTR;                       // [BN][LSTR]
  _Float16* Bl = Bh + BN * LSTR;

  const int tid  = threadIdx.x;
  const int wave = tid >> 5;
  const int lane = tid & 31;
  const int lh   = lane >> 4;     // lane half (K-group select)
  const int ln   = lane & 15;     // M row (A) / N col (B/C)

#ifdef USE_TDM
  // ---- TDM staging: wave 0 issues 4 DMA descriptors, waits TENSORcnt ----
  if (tid < 32) {
    const unsigned int shbase = (unsigned int)(uintptr_t)sh;
    tdm_load_tile_f16(shbase,                    Yh + (size_t)blockIdx.x * BM * HIDDEN, BM);
    tdm_load_tile_f16(shbase + A_SZ,             Yl + (size_t)blockIdx.x * BM * HIDDEN, BM);
    tdm_load_tile_f16(shbase + 2 * A_SZ,         Wh + (size_t)blockIdx.y * BN * HIDDEN, BN);
    tdm_load_tile_f16(shbase + 2 * A_SZ + B_SZ,  Wl + (size_t)blockIdx.y * BN * HIDDEN, BN);
    __builtin_amdgcn_s_wait_tensorcnt(0);
  }
#else
  // ---- fallback: manual vectorized copy into padded LDS ----
  {
    const _Float16* ah = Yh + (size_t)blockIdx.x * BM * HIDDEN;
    const _Float16* al = Yl + (size_t)blockIdx.x * BM * HIDDEN;
    for (int idx = tid * 8; idx < BM * HIDDEN; idx += 256 * 8) {
      const int r = idx >> 8, k = idx & 255;
      *reinterpret_cast<v8h*>(&Ah[r * LSTR + k]) = *reinterpret_cast<const v8h*>(ah + idx);
      *reinterpret_cast<v8h*>(&Al[r * LSTR + k]) = *reinterpret_cast<const v8h*>(al + idx);
    }
    const _Float16* bh = Wh + (size_t)blockIdx.y * BN * HIDDEN;
    const _Float16* bl = Wl + (size_t)blockIdx.y * BN * HIDDEN;
    for (int idx = tid * 8; idx < BN * HIDDEN; idx += 256 * 8) {
      const int r = idx >> 8, k = idx & 255;
      *reinterpret_cast<v8h*>(&Bh[r * LSTR + k]) = *reinterpret_cast<const v8h*>(bh + idx);
      *reinterpret_cast<v8h*>(&Bl[r * LSTR + k]) = *reinterpret_cast<const v8h*>(bl + idx);
    }
  }
#endif
  __syncthreads();

  v8f acc[NT];
#pragma unroll
  for (int nt = 0; nt < NT; ++nt) acc[nt] = (v8f)0.0f;

  const _Float16* arow_h = Ah + (wave * 16 + ln) * LSTR;
  const _Float16* arow_l = Al + (wave * 16 + ln) * LSTR;

  for (int kc = 0; kc < HIDDEN; kc += 32) {
    // A fragment per documented 16-bit A layout:
    //   lanes 0-15: K = kc+{0..7, 16..23}; lanes 16-31: K = kc+{8..15, 24..31}
    const int kA = kc + 8 * lh;
    AFrag ah, al;
    ah.q[0] = *reinterpret_cast<const v4h*>(arow_h + kA + 0);
    ah.q[1] = *reinterpret_cast<const v4h*>(arow_h + kA + 4);
    ah.q[2] = *reinterpret_cast<const v4h*>(arow_h + kA + 16);
    ah.q[3] = *reinterpret_cast<const v4h*>(arow_h + kA + 20);
    al.q[0] = *reinterpret_cast<const v4h*>(arow_l + kA + 0);
    al.q[1] = *reinterpret_cast<const v4h*>(arow_l + kA + 4);
    al.q[2] = *reinterpret_cast<const v4h*>(arow_l + kA + 16);
    al.q[3] = *reinterpret_cast<const v4h*>(arow_l + kA + 20);

    const int kB = kc + 8 * lh;    // B: lanes 0-15 K=kc+{0..7,16..23}, 16-31 K=kc+{8..15,24..31}
#pragma unroll
    for (int nt = 0; nt < NT; ++nt) {
      const _Float16* brow_h = Bh + (nt * 16 + ln) * LSTR + kB;
      const _Float16* brow_l = Bl + (nt * 16 + ln) * LSTR + kB;
      BFrag bh, bl;
      bh.o[0] = *reinterpret_cast<const v8h*>(brow_h);
      bh.o[1] = *reinterpret_cast<const v8h*>(brow_h + 16);
      bl.o[0] = *reinterpret_cast<const v8h*>(brow_l);
      bl.o[1] = *reinterpret_cast<const v8h*>(brow_l + 16);

      acc[nt] = __builtin_amdgcn_wmma_f32_16x16x32_f16(
          false, ah.v, false, bh.v, (short)0, acc[nt], false, false);
      acc[nt] = __builtin_amdgcn_wmma_f32_16x16x32_f16(
          false, ah.v, false, bl.v, (short)0, acc[nt], false, false);
      acc[nt] = __builtin_amdgcn_wmma_f32_16x16x32_f16(
          false, al.v, false, bh.v, (short)0, acc[nt], false, false);
    }
  }

  // Epilogue: C layout -> VGPR r holds (M = r + 8*lh, N = ln) per lane half.
  const int mbase = blockIdx.x * BM + wave * 16 + 8 * lh;
#pragma unroll
  for (int nt = 0; nt < NT; ++nt) {
    const int col = blockIdx.y * BN + nt * 16 + ln;
    const float bias = b_out[col];
#pragma unroll
    for (int r = 0; r < 8; ++r) {
      logits[(size_t)(mbase + r) * VOCAB + col] = acc[nt][r] + bias;
    }
  }
}

// ---------------------------------------------------------------------------
// Host launcher
// ---------------------------------------------------------------------------
extern "C" void kernel_launch(void* const* d_in, const int* in_sizes, int n_in,
                              void* d_out, int out_size, void* d_ws, size_t ws_size,
                              hipStream_t stream) {
  const int*   inputs = (const int*)  d_in[0];
  const float* state  = (const float*)d_in[1];
  const float* W_ih   = (const float*)d_in[2];
  const float* b_ih   = (const float*)d_in[3];
  const float* W_hh   = (const float*)d_in[4];
  const float* b_hh   = (const float*)d_in[5];
  const float* W_out  = (const float*)d_in[6];
  const float* b_out  = (const float*)d_in[7];

  float* logits = (float*)d_out;                       // ROWS * VOCAB
  float* h_last = logits + (size_t)ROWS * VOCAB;       // BATCH * HIDDEN

  // workspace layout (bytes): xproj f32 | Yh | Yl | Wh | Wl  (~25.8 MB)
  char* ws = (char*)d_ws;
  float*    xproj = (float*)ws;                                   // 8 MB
  _Float16* Yh    = (_Float16*)(ws + 8388608);                    // 4 MB
  _Float16* Yl    = (_Float16*)(ws + 8388608 + 4194304);          // 4 MB
  _Float16* Wh    = (_Float16*)(ws + 8388608 + 2 * 4194304);      // 5 MB
  _Float16* Wl    = (_Float16*)(ws + 8388608 + 2 * 4194304 + 5120000);

  // 1) embedding gather + bias
  rnn_gather<<<ROWS, 256, 0, stream>>>(inputs, W_ih, b_ih, b_hh, xproj);

  // 1b) one-shot W_out hi/lo split (independent of k2)
  wout_split<<<(VOCAB * HIDDEN) / (256 * 4), 256, 0, stream>>>(W_out, Wh, Wl);

  // 2) recurrence: 32 independent batch rows, W_hh resident in LDS
  const size_t lds2 = (size_t)(HIDDEN * WSTRIDE + 2 * HIDDEN) * sizeof(float);
  rnn_recurrence<<<BATCH, 256, lds2, stream>>>(xproj, state, W_hh, Yh, Yl, h_last);

  // 3) output GEMM with split-f16 WMMA + TDM staging: 64 x 125 blocks
  const size_t lds3 = (size_t)(2 * A_SZ + 2 * B_SZ);   // 219648 B
  dim3 grid3(ROWS / BM, VOCAB / BN);
  rnn_logits_gemm<<<grid3, 256, lds3, stream>>>(Yh, Yl, Wh, Wl, b_out, logits);
}
